// MoE_20401094656194
// MI455X (gfx1250) — matrix operand order repaired and checked by reading, not compiled, not run on hardware
//
#include <hip/hip_runtime.h>
#include <hip/hip_bf16.h>

// ---------------------------------------------------------------------------
// MoE (gate -> top-2 routed SwiGLU experts + shared SwiGLU expert) for gfx1250
// fp32 inputs; compute via V_WMMA_F32_16X16X32_BF16.
//  - fp32 weights staged cooperatively through LDS (converted to bf16 once per
//    block), double-buffered 64-K stages, one barrier per 16 WMMAs.
//  - B fragments in distinct registers so ds_loads pipeline (partial dscnt waits).
//  - global_prefetch for the stage-after-next weight tile.
//  - expert dispatch via bucketed gather (only top-2 work computed).
// ---------------------------------------------------------------------------

typedef __bf16 bf16;
typedef __attribute__((ext_vector_type(16))) __bf16 v16bf;
typedef __attribute__((ext_vector_type(8)))  __bf16 v8bf;
typedef __attribute__((ext_vector_type(4)))  __bf16 v4bf;
typedef __attribute__((ext_vector_type(8)))  float  v8f;

#define T_TOK 3136   // B*S = 16*196
#define DIM_  768
#define NEXP  16
#define MI_   2048
#define SI_   4096

// LDS layout: per staged matrix, 64 columns x 64 K-values (bf16), column
// stride padded to 72 halfs (144 B, 16B-aligned, spreads banks).
#define LDS_CSTRIDE 72

// ---- WMMA fragment loaders (contiguous K order, same in A and B, wave32) --------------------
// A (16x32 bf16): lane L -> row M = L%16; elems 0..15 = K[(L/16)*16 .. +15].
static __device__ __forceinline__ v16bf load_a_bf16(const bf16* rowPtr, int kk, int khalf) {
  const v8bf lo = *(const v8bf*)(rowPtr + kk + khalf * 16);
  const v8bf hi = *(const v8bf*)(rowPtr + kk + khalf * 16 + 8);
  return __builtin_shufflevector(lo, hi, 0,1,2,3,4,5,6,7,8,9,10,11,12,13,14,15);
}

// B (32x16 bf16): lane L -> col N = L%16; elems 0..15 = K[(L/16)*16 .. +15].
// Fragment read from LDS (two ds_load_b128). kc selects 32-K chunk within stage.
static __device__ __forceinline__ v16bf lds_b_frag(const bf16* mat, int ct, int m,
                                                   int khalf, int kc) {
  const bf16* p = mat + (ct * 16 + m) * LDS_CSTRIDE + kc * 32 + khalf * 16;
  const v8bf lo = *(const v8bf*)(p);
  const v8bf hi = *(const v8bf*)(p + 8);
  return __builtin_shufflevector(lo, hi, 0,1,2,3,4,5,6,7,8,9,10,11,12,13,14,15);
}

// ---- cooperative B staging: 8 fp32 K-values per thread per chunk -----------
struct F8 { float4 a, b; };

static __device__ __forceinline__ F8 gload8(const float* colPtr, int kk, int kq) {
  const float4* p = (const float4*)(colPtr + kk + kq * 8);
  F8 r; r.a = p[0]; r.b = p[1];
  return r;
}

static __device__ __forceinline__ void sstore8(bf16* colDst, int kq, F8 f) {
  v8bf o;
  o[0] = (bf16)f.a.x; o[1] = (bf16)f.a.y; o[2] = (bf16)f.a.z; o[3] = (bf16)f.a.w;
  o[4] = (bf16)f.b.x; o[5] = (bf16)f.b.y; o[6] = (bf16)f.b.z; o[7] = (bf16)f.b.w;
  *(v8bf*)(colDst + kq * 8) = o;
}

// ---- fp32 -> bf16 activation conversion ------------------------------------
__global__ void __launch_bounds__(256)
k_cvt(const float* __restrict__ x, bf16* __restrict__ xb, int n4) {
  const int i = blockIdx.x * 256 + threadIdx.x;
  if (i >= n4) return;
  const float4 f = ((const float4*)x)[i];
  v4bf o;
  o[0] = (bf16)f.x; o[1] = (bf16)f.y; o[2] = (bf16)f.z; o[3] = (bf16)f.w;
  ((v4bf*)xb)[i] = o;
}

// ---- gate: softmax scores, top-2, bucket append ----------------------------
__global__ void __launch_bounds__(256)
k_gate(const float* __restrict__ x, const float* __restrict__ gw,
       int* __restrict__ counts, int* __restrict__ btok,
       int* __restrict__ bslot, float* __restrict__ bw) {
  const int t = blockIdx.x * 8 + (threadIdx.x >> 5);   // one wave32 per token
  if (t >= T_TOK) return;
  const int lane = threadIdx.x & 31;
  const float* xr = x + (size_t)t * DIM_;

  float acc[NEXP];
#pragma unroll
  for (int e = 0; e < NEXP; ++e) acc[e] = 0.0f;
  for (int d = lane; d < DIM_; d += 32) {
    const float xv = xr[d];
#pragma unroll
    for (int e = 0; e < NEXP; ++e) acc[e] += xv * gw[e * DIM_ + d];
  }
#pragma unroll
  for (int e = 0; e < NEXP; ++e) {
#pragma unroll
    for (int off = 16; off > 0; off >>= 1)
      acc[e] += __shfl_xor(acc[e], off, 32);
  }
  if (lane == 0) {
    float m = acc[0];
#pragma unroll
    for (int e = 1; e < NEXP; ++e) m = fmaxf(m, acc[e]);
    float p[NEXP]; float den = 0.0f;
#pragma unroll
    for (int e = 0; e < NEXP; ++e) { p[e] = __expf(acc[e] - m); den += p[e]; }
    const float inv = 1.0f / den;        // ROUTE_SCALE == 1.0
    int i0 = 0; float v0 = p[0];
#pragma unroll
    for (int e = 1; e < NEXP; ++e) if (p[e] > v0) { v0 = p[e]; i0 = e; }
    int i1 = (i0 == 0) ? 1 : 0; float v1 = p[i1];
#pragma unroll
    for (int e = 0; e < NEXP; ++e) if (e != i0 && p[e] > v1) { v1 = p[e]; i1 = e; }
    const int p0 = atomicAdd(&counts[i0], 1);
    btok[i0 * T_TOK + p0] = t; bslot[i0 * T_TOK + p0] = t * 2;     bw[i0 * T_TOK + p0] = v0 * inv;
    const int p1 = atomicAdd(&counts[i1], 1);
    btok[i1 * T_TOK + p1] = t; bslot[i1 * T_TOK + p1] = t * 2 + 1; bw[i1 * T_TOK + p1] = v1 * inv;
  }
}

// ---- up-projection: H = silu(X @ W1^T) * (X @ W3^T), gathered rows ---------
// Block: 8 waves = 128 rows x 64 cols. W1/W3 64-K tiles staged via LDS.
__global__ void __launch_bounds__(256)
k_up(const bf16* __restrict__ X, const float* __restrict__ W1,
     const float* __restrict__ W3, bf16* __restrict__ H,
     const int* __restrict__ counts, const int* __restrict__ btok,
     const int* __restrict__ bslot, int K, int inter, int rowsTotal) {
  __shared__ __align__(16) bf16 sB[2][2][64 * LDS_CSTRIDE];  // 36 KB

  const int e    = blockIdx.z;
  const int tid  = threadIdx.x;
  const int lane = tid & 31;
  const int wav  = tid >> 5;
  const int n    = btok ? counts[e] : rowsTotal;
  if (blockIdx.y * 128 >= n) return;                 // block-uniform exit
  const int rowBase = blockIdx.y * 128 + wav * 16;
  const int colBase = blockIdx.x * 64;
  const int m     = lane & 15;
  const int khalf = lane >> 4;

  const int r  = rowBase + m;
  const int rc = (r < n) ? r : (n - 1);
  const int aRow = btok ? btok[e * rowsTotal + rc] : rc;
  const bf16* aPtr = X + (size_t)aRow * K;

  // staging role: thread -> (column, K-quad); 4 threads per column cover 64 K
  const int scol = tid >> 2;                          // 0..63
  const int kq   = tid & 3;                           // 0..3
  const size_t wOff = (size_t)e * inter * K;
  const float* w1c = W1 + wOff + (size_t)(colBase + scol) * K;
  const float* w3c = W3 + wOff + (size_t)(colBase + scol) * K;

  F8 f1a = gload8(w1c, 0, kq), f1b = gload8(w1c, 32, kq);
  F8 f3a = gload8(w3c, 0, kq), f3b = gload8(w3c, 32, kq);

  const v8f vz = {0.f,0.f,0.f,0.f,0.f,0.f,0.f,0.f};
  v8f acc1[4], acc3[4];
#pragma unroll
  for (int ct = 0; ct < 4; ++ct) { acc1[ct] = vz; acc3[ct] = vz; }

  const int steps = K / 64;
  for (int i = 0; i < steps; ++i) {
    const int s = i & 1;
    bf16* c1 = &sB[s][0][scol * LDS_CSTRIDE];
    bf16* c3 = &sB[s][1][scol * LDS_CSTRIDE];
    sstore8(c1, kq, f1a); sstore8(c1 + 32, kq, f1b);
    sstore8(c3, kq, f3a); sstore8(c3 + 32, kq, f3b);
    if (i + 1 < steps) {                              // register prefetch
      const int kk = (i + 1) * 64;
      f1a = gload8(w1c, kk, kq); f1b = gload8(w1c, kk + 32, kq);
      f3a = gload8(w3c, kk, kq); f3b = gload8(w3c, kk + 32, kq);
    }
    if (i + 2 < steps) {                              // L2 warm for stage+2
      __builtin_prefetch(w1c + (i + 2) * 64, 0, 3);
      __builtin_prefetch(w3c + (i + 2) * 64, 0, 3);
    }
    __syncthreads();                                  // stage s visible
#pragma unroll
    for (int kc = 0; kc < 2; ++kc) {
      const v16bf a = load_a_bf16(aPtr, i * 64 + kc * 32, khalf);
      v16bf b1[4], b3[4];
#pragma unroll
      for (int ct = 0; ct < 4; ++ct) b1[ct] = lds_b_frag(&sB[s][0][0], ct, m, khalf, kc);
#pragma unroll
      for (int ct = 0; ct < 4; ++ct) b3[ct] = lds_b_frag(&sB[s][1][0], ct, m, khalf, kc);
#pragma unroll
      for (int ct = 0; ct < 4; ++ct) {
        acc1[ct] = __builtin_amdgcn_wmma_f32_16x16x32_bf16(false, a, false, b1[ct],
                                                           (short)0, acc1[ct], false, false);
        acc3[ct] = __builtin_amdgcn_wmma_f32_16x16x32_bf16(false, a, false, b3[ct],
                                                           (short)0, acc3[ct], false, false);
      }
    }
    // double buffer: next iteration's barrier (with its s_wait_dscnt)
    // protects stage reuse two iterations later.
  }

#pragma unroll
  for (int i = 0; i < 8; ++i) {
    const int row = rowBase + i + khalf * 8;          // C layout: VGPR i -> M=i (+8 hi lanes)
    if (row < n) {
      const int outRow = bslot ? bslot[e * rowsTotal + row] : row;
      bf16* hr = H + (size_t)outRow * inter;
#pragma unroll
      for (int ct = 0; ct < 4; ++ct) {
        const float g = acc1[ct][i];
        const float s = g / (1.0f + __expf(-g));      // silu
        hr[colBase + ct * 16 + m] = (bf16)(s * acc3[ct][i]);
      }
    }
  }
}

// ---- down-projection: out[token] += weight * (H @ W2^T) --------------------
__global__ void __launch_bounds__(256)
k_down(const bf16* __restrict__ Hm, const float* __restrict__ W2,
       float* __restrict__ out,
       const int* __restrict__ counts, const int* __restrict__ btok,
       const int* __restrict__ bslot, const float* __restrict__ bw,
       int K, int D, int rowsTotal) {
  __shared__ __align__(16) bf16 sB[2][64 * LDS_CSTRIDE];     // 18 KB

  const int e    = blockIdx.z;
  const int tid  = threadIdx.x;
  const int lane = tid & 31;
  const int wav  = tid >> 5;
  const int n    = btok ? counts[e] : rowsTotal;
  if (blockIdx.y * 128 >= n) return;
  const int rowBase = blockIdx.y * 128 + wav * 16;
  const int colBase = blockIdx.x * 64;
  const int m     = lane & 15;
  const int khalf = lane >> 4;

  const int r  = rowBase + m;
  const int rc = (r < n) ? r : (n - 1);
  const int aRow = bslot ? bslot[e * rowsTotal + rc] : rc;
  const bf16* aPtr = Hm + (size_t)aRow * K;

  const int scol = tid >> 2;
  const int kq   = tid & 3;
  const float* w2c = W2 + (size_t)e * D * K + (size_t)(colBase + scol) * K;

  F8 f2a = gload8(w2c, 0, kq), f2b = gload8(w2c, 32, kq);

  const v8f vz = {0.f,0.f,0.f,0.f,0.f,0.f,0.f,0.f};
  v8f acc[4];
#pragma unroll
  for (int ct = 0; ct < 4; ++ct) acc[ct] = vz;

  const int steps = K / 64;
  for (int i = 0; i < steps; ++i) {
    const int s = i & 1;
    bf16* c2 = &sB[s][scol * LDS_CSTRIDE];
    sstore8(c2, kq, f2a); sstore8(c2 + 32, kq, f2b);
    if (i + 1 < steps) {
      const int kk = (i + 1) * 64;
      f2a = gload8(w2c, kk, kq); f2b = gload8(w2c, kk + 32, kq);
    }
    if (i + 2 < steps) __builtin_prefetch(w2c + (i + 2) * 64, 0, 3);
    __syncthreads();
#pragma unroll
    for (int kc = 0; kc < 2; ++kc) {
      const v16bf a = load_a_bf16(aPtr, i * 64 + kc * 32, khalf);
      v16bf b[4];
#pragma unroll
      for (int ct = 0; ct < 4; ++ct) b[ct] = lds_b_frag(&sB[s][0], ct, m, khalf, kc);
#pragma unroll
      for (int ct = 0; ct < 4; ++ct)
        acc[ct] = __builtin_amdgcn_wmma_f32_16x16x32_bf16(false, a, false, b[ct],
                                                          (short)0, acc[ct], false, false);
    }
  }

#pragma unroll
  for (int i = 0; i < 8; ++i) {
    const int row = rowBase + i + khalf * 8;
    if (row < n) {
      const int   tok = btok ? btok[e * rowsTotal + row] : row;
      const float wgt = bw   ? bw[e * rowsTotal + row]   : 1.0f;
      float* orow = out + (size_t)tok * D;
#pragma unroll
      for (int ct = 0; ct < 4; ++ct)
        atomicAdd(&orow[colBase + ct * 16 + m], acc[ct][i] * wgt);
    }
  }
}

// ---------------------------------------------------------------------------
extern "C" void kernel_launch(void* const* d_in, const int* in_sizes, int n_in,
                              void* d_out, int out_size, void* d_ws, size_t ws_size,
                              hipStream_t stream) {
  (void)in_sizes; (void)n_in; (void)ws_size;
  const float* x   = (const float*)d_in[0];
  const float* gw  = (const float*)d_in[1];
  const float* w1  = (const float*)d_in[2];
  const float* w2  = (const float*)d_in[3];
  const float* w3  = (const float*)d_in[4];
  const float* ws1 = (const float*)d_in[5];
  const float* ws2 = (const float*)d_in[6];
  const float* ws3 = (const float*)d_in[7];
  float* out = (float*)d_out;

  char* base = (char*)d_ws;
  size_t off = 0;
  auto carve = [&](size_t bytes) -> void* {
    void* p = base + off;
    off = (off + bytes + 255) & ~(size_t)255;
    return p;
  };
  bf16*  Xb     = (bf16*) carve((size_t)T_TOK * DIM_ * sizeof(bf16));
  int*   counts = (int*)  carve(NEXP * sizeof(int));
  int*   btok   = (int*)  carve((size_t)NEXP * T_TOK * sizeof(int));
  int*   bslot  = (int*)  carve((size_t)NEXP * T_TOK * sizeof(int));
  float* bw     = (float*)carve((size_t)NEXP * T_TOK * sizeof(float));
  bf16*  Hr     = (bf16*) carve((size_t)(2 * T_TOK) * MI_ * sizeof(bf16));
  bf16*  Hs     = (bf16*) carve((size_t)T_TOK * SI_ * sizeof(bf16));

  hipMemsetAsync(counts, 0, NEXP * sizeof(int), stream);
  hipMemsetAsync(out, 0, (size_t)out_size * sizeof(float), stream);

  k_cvt<<<(T_TOK * DIM_ / 4 + 255) / 256, 256, 0, stream>>>(x, Xb, T_TOK * DIM_ / 4);
  k_gate<<<T_TOK / 8, 256, 0, stream>>>(x, gw, counts, btok, bslot, bw);

  const int rowTiles = (T_TOK + 127) / 128;
  // routed experts (top-2 dispatch)
  dim3 gUpR(MI_ / 64, rowTiles, NEXP);
  k_up<<<gUpR, 256, 0, stream>>>(Xb, w1, w3, Hr, counts, btok, bslot, DIM_, MI_, T_TOK);
  dim3 gDnR(DIM_ / 64, rowTiles, NEXP);
  k_down<<<gDnR, 256, 0, stream>>>(Hr, w2, out, counts, btok, bslot, bw, MI_, DIM_, T_TOK);

  // shared expert (dense over all tokens)
  dim3 gUpS(SI_ / 64, rowTiles, 1);
  k_up<<<gUpS, 256, 0, stream>>>(Xb, ws1, ws3, Hs, nullptr, nullptr, nullptr, DIM_, SI_, T_TOK);
  dim3 gDnS(DIM_ / 64, rowTiles, 1);
  k_down<<<gDnS, 256, 0, stream>>>(Hs, ws2, out, nullptr, nullptr, nullptr, nullptr, SI_, DIM_, T_TOK);
}
